// C_Aggregation_24807731101830
// MI455X (gfx1250) — matrix operand order repaired and checked by reading, not compiled, not run on hardware
//
#include <hip/hip_runtime.h>

// ---------------------------------------------------------------------------
// C_Aggregation on MI455X (gfx1250, wave32).
//  Phase 1: patch-embed conv as f16 WMMA GEMM (f32 accumulate).
//           Block = 8 waves sharing one 64-channel weight panel staged in LDS
//           (two 384-wide K phases, ~49KB, padded rows -> conflict-free-ish).
//           Wave tile = 32 positions x 64 channels = 8 wmma / K-step.
//  Phase 2: per-(b,channel)-column sequential smoothing scan (columns are
//           fully independent -> 1 thread/column, coalesced, L2-resident)
//  Phase 3: reshape/crop gather to output
// Workspace layout (needs ~83.2 MB):
//   [0)                 buf : float [1156][12288]   (s-major, col = b*768+c1)
//   [56,807,424)        Ah  : f16   [16][1024][768] (A fragments, K-major)
//   [81,973,248)        Wh  : f16   [768][768]      (B fragments, K-major)
// ---------------------------------------------------------------------------

typedef __attribute__((ext_vector_type(16))) _Float16 v16h;
typedef __attribute__((ext_vector_type(8)))  _Float16 v8h;
typedef __attribute__((ext_vector_type(8)))  float    v8f;

#define EMB     768
#define BATCH   16
#define NPOS    1156              // 34*34 flattened patch grid (with pad ring)
#define TCOLS   (BATCH * EMB)     // 12288 independent scan columns
#define AH_OFF  56807424u         // 1156*12288*4 - 12288 (last 12288 B of buf overlap Ah)
#define WH_OFF  81973248u         // + 16*1024*768*2
#define KS      384               // K-phase width staged in LDS
#define BPAD    8                 // row pad (halves): stride 392h = 196 dw,
                                  // 196 % 64 = 4 banks/lane -> no 16-way hit

// ---- Phase 0a: w (already [oc][k] row-major, k=(c,py,px)) -> f16 -----------
__global__ void prep_w(const float* __restrict__ w, _Float16* __restrict__ Wh) {
    int idx = blockIdx.x * blockDim.x + threadIdx.x;   // 768*768 exact
    Wh[idx] = (_Float16)w[idx];
}

// ---- Phase 0b: stage x patches as A rows: Ah[b][pos][k], k=(c,py,px) -------
__global__ void prep_a(const float* __restrict__ x, _Float16* __restrict__ Ah) {
    int idx = blockIdx.x * blockDim.x + threadIdx.x;   // 16*1024*768 exact
    int k   = idx % EMB;
    int pos = (idx / EMB) & 1023;
    int b   = idx / (EMB * 1024);
    int c  = k >> 8, py = (k >> 4) & 15, px = k & 15;
    int ph = pos >> 5, pw = pos & 31;
    size_t xi = (((size_t)(b * 3 + c)) * 512 + (ph * 16 + py)) * 512 + (pw * 16 + px);
    Ah[idx] = (_Float16)x[xi];                          // coalesced in/out
}

// ---- Phase 0c: whole buffer = bias broadcast (pad ring stays bias-only) ----
__global__ void init_bias(const float* __restrict__ bias, float* __restrict__ buf) {
    int idx = blockIdx.x * blockDim.x + threadIdx.x;   // 1156*12288 exact
    buf[idx] = bias[idx % EMB];                        // 12288 % 768 == 0
}

// ---- Phase 1: GEMM. Block = 8 waves x (32 pos x 64 oc), LDS-staged B -------
__global__ __launch_bounds__(256) void patch_gemm(
        const _Float16* __restrict__ Ah, const _Float16* __restrict__ Wh,
        const float* __restrict__ bias, float* __restrict__ buf) {
    __shared__ _Float16 Bs[64][KS + BPAD];             // ~49 KB

    const int tid  = threadIdx.x;
    const int lane = tid & 31;
    const int wv   = tid >> 5;                         // wave in block (0..7)
    // 768 blocks = b(16) x pgrp(4) x ocgrp(12); wave picks ptile = pgrp*8+wv
    const int b   = blockIdx.x / 48;
    const int rem = blockIdx.x % 48;
    const int pg  = rem / 12;
    const int og  = rem % 12;
    const int pt  = pg * 8 + wv;                       // 0..31 == ph row
    const int p0  = pt << 5;                           // 32 positions (one row)
    const int oc0 = og * 64;
    const int col  = lane & 15;                        // N (oc within 16)
    const int half = lane >> 4;                        // K-half per ISA layout

    v8f acc0[4], acc1[4];
    #pragma unroll
    for (int n = 0; n < 4; ++n) { acc0[n] = (v8f){}; acc1[n] = (v8f){}; }

    // A frag rows: lane holds M=col; K chunks {half*8..+8} and {16+half*8..+8}
    const _Float16* Ab0 = Ah + (size_t)(b * 1024 + p0 +      col) * EMB + half * 8;
    const _Float16* Ab1 = Ah + (size_t)(b * 1024 + p0 + 16 + col) * EMB + half * 8;
    const _Float16* Wp  = Wh + (size_t)oc0 * EMB;

    for (int kp = 0; kp < 2; ++kp) {                   // two K phases of 384
        const int kbase = kp * KS;
        __syncthreads();                               // previous-phase readers done
        // cooperative stage: 64 rows x 384 halves = 3072 v8h chunks
        for (int c = tid; c < 3072; c += 256) {
            int row = c / 48;                          // 48 chunks per row
            int off = (c % 48) * 8;
            *(v8h*)&Bs[row][off] =
                *(const v8h*)(Wp + (size_t)row * EMB + kbase + off);
        }
        __syncthreads();

        for (int kk = 0; kk < KS; kk += 32) {
            const int k0 = kbase + kk;
            __builtin_prefetch(Ab0 + k0 + 128, 0, 1);  // global_prefetch_b8
            __builtin_prefetch(Ab1 + k0 + 128, 0, 1);
            v8h a0lo = *(const v8h*)(Ab0 + k0);
            v8h a0hi = *(const v8h*)(Ab0 + k0 + 16);
            v8h a1lo = *(const v8h*)(Ab1 + k0);
            v8h a1hi = *(const v8h*)(Ab1 + k0 + 16);
            v16h a0 = __builtin_shufflevector(a0lo, a0hi,
                      0,1,2,3,4,5,6,7,8,9,10,11,12,13,14,15);
            v16h a1 = __builtin_shufflevector(a1lo, a1hi,
                      0,1,2,3,4,5,6,7,8,9,10,11,12,13,14,15);
            #pragma unroll
            for (int n = 0; n < 4; ++n) {
                // B frag from LDS: lane = N=col, K chunks {half*8..+8} and {16+half*8..+8}
                const _Float16* bp = &Bs[n * 16 + col][kk + half * 8];
                v8h blo = *(const v8h*)(bp);
                v8h bhi = *(const v8h*)(bp + 16);
                v16h bf = __builtin_shufflevector(blo, bhi,
                          0,1,2,3,4,5,6,7,8,9,10,11,12,13,14,15);
                acc0[n] = __builtin_amdgcn_wmma_f32_16x16x32_f16(
                          false, a0, false, bf, (short)0, acc0[n], false, false);
                acc1[n] = __builtin_amdgcn_wmma_f32_16x16x32_f16(
                          false, a1, false, bf, (short)0, acc1[n], false, false);
            }
        }
    }

    const int sBase = (pt + 1) * 34 + 1;               // pad-ring shifted row
    const size_t cb = (size_t)b * EMB + oc0 + col;
    #pragma unroll
    for (int n = 0; n < 4; ++n) {
        const float bv = bias[oc0 + n * 16 + col];
        #pragma unroll
        for (int r = 0; r < 8; ++r) {
            int m = r + half * 8;                      // M per C/D ISA layout
            buf[(size_t)(sBase + m     ) * TCOLS + cb + n * 16] = acc0[n][r] + bv;
            buf[(size_t)(sBase + m + 16) * TCOLS + cb + n * 16] = acc1[n][r] + bv;
        }
    }
}

// ---- Phase 2: sequential scan; one thread per independent (b,c1) column ----
__global__ void smooth_scan(float* __restrict__ buf) {
    const int t = blockIdx.x * blockDim.x + threadIdx.x;
    if (t >= TCOLS) return;
    const long S = TCOLS;
    for (int i = 1; i <= 30; ++i) {
        const long base = (long)i * 32;                // stride-32 flat idx
        float a0 = buf[(base - 32) * S + t];           // row i-1 (updated)
        float a1 = buf[(base - 31) * S + t];
        float a2 = buf[(base - 30) * S + t];
        float b0 = buf[(base     ) * S + t];           // row i (orig at j=0)
        float b2 = buf[(base +  2) * S + t];
        float c0 = buf[(base + 32) * S + t];           // row i+1 (orig)
        float c1v= buf[(base + 33) * S + t];
        float c2 = buf[(base + 34) * S + t];
        for (int j = 1; j <= 31; ++j) {
            float o = (a0 + a1 + a2 + b0 + b2 + c0 + c1v + c2) * 0.125f;
            buf[(base + j) * S + t] = o;
            a0 = a1; a1 = a2; a2 = buf[(base - 30 + j) * S + t];
            b0 = o;           b2 = buf[(base +  2 + j) * S + t];
            c0 = c1v; c1v = c2; c2 = buf[(base + 34 + j) * S + t];
        }
    }
}

// ---- Phase 3: reshape (c1*1156+s -> c,h,w) + crop gather -------------------
__global__ void gather_out(const float* __restrict__ buf, float* __restrict__ out) {
    int idx = blockIdx.x * blockDim.x + threadIdx.x;   // 16*3*512*512 exact
    int wv = idx & 511;
    int h  = (idx >> 9) & 511;
    int c  = (idx / 262144) % 3;
    int b  = idx / 786432;
    int f  = c * 295936 + (h + 16) * 544 + (wv + 16);  // flat within batch
    int c1 = f / 1156;
    int s  = f - c1 * 1156;
    out[idx] = buf[(size_t)s * TCOLS + (size_t)b * EMB + c1];
}

extern "C" void kernel_launch(void* const* d_in, const int* in_sizes, int n_in,
                              void* d_out, int out_size, void* d_ws, size_t ws_size,
                              hipStream_t stream) {
    const float* x    = (const float*)d_in[0];   // (16,3,512,512)
    const float* w    = (const float*)d_in[1];   // (768,3,16,16)
    const float* bias = (const float*)d_in[2];   // (768,)
    float* out = (float*)d_out;

    char* ws = (char*)d_ws;
    float*    buf = (float*)ws;
    _Float16* Ah  = (_Float16*)(ws + AH_OFF);
    _Float16* Wh  = (_Float16*)(ws + WH_OFF);

    prep_w    <<< 589824 / 256, 256, 0, stream>>>(w, Wh);
    prep_a    <<< 12582912 / 256, 256, 0, stream>>>(x, Ah);
    init_bias <<< (NPOS * TCOLS) / 256, 256, 0, stream>>>(bias, buf);
    patch_gemm<<< 768, 256, 0, stream>>>(Ah, Wh, bias, buf);  // 6144 waves
    smooth_scan<<< (TCOLS + 255) / 256, 256, 0, stream>>>(buf);
    gather_out<<< 12582912 / 256, 256, 0, stream>>>(buf, out);
}
